// LSTMFrame_80032420594213
// MI455X (gfx1250) — matrix operand order repaired and checked by reading, not compiled, not run on hardware
//
#include <hip/hip_runtime.h>

// ---------------------------------------------------------------------------
// Bidirectional 2-layer LSTM for MI455X (gfx1250), bf16 WMMA + fp32 accum.
// T=512, B=64, I=512, H=512, G=4H=2048.
//
// Recurrent scan: 16 WGPs per direction; each WGP keeps its 128KB Whh slice
// resident in LDS for all 512 timesteps (zero weight re-streaming), exchanges
// the 64KB hidden state through a double-buffered L2 buffer with a per-
// direction inter-block barrier.
// ---------------------------------------------------------------------------

#define TT 512
#define BB 64
#define HH 512
#define GG 2048

typedef __attribute__((ext_vector_type(16))) __bf16 v16bf;
typedef __attribute__((ext_vector_type(8)))  float  v8f;

union ABFrag { uint4 u[2]; v16bf v; };

// A fragment (16-bit A, 16x32 MxK): lane holds row M=lane%16; elements e=0..7
// are K = half*8 + e, e=8..15 are K = 16 + half*8 + (e-8)  (half = lane/16).
__device__ __forceinline__ v16bf load_a_frag(const __bf16* row, int kbase, int half) {
  ABFrag f;
  const __bf16* p = row + kbase + half * 8;
  f.u[0] = *reinterpret_cast<const uint4*>(p);
  f.u[1] = *reinterpret_cast<const uint4*>(p + 16);
  return f.v;
}

// B fragment from global, uniform base + 32-bit byte offset (SADDR form).
__device__ __forceinline__ v16bf load_b_g(const __bf16* wp, unsigned byteOff) {
  ABFrag f;
  const char* p = (const char*)wp + byteOff;
  f.u[0] = *reinterpret_cast<const uint4*>(p);
  f.u[1] = *reinterpret_cast<const uint4*>(p + 16);
  return f.v;
}

// B fragment from LDS (element offset), lane's 16 bf16 contiguous.
__device__ __forceinline__ v16bf load_b_lds(const __bf16* base, unsigned elemOff) {
  ABFrag f;
  const __bf16* p = base + elemOff;
  f.u[0] = *reinterpret_cast<const uint4*>(p);
  f.u[1] = *reinterpret_cast<const uint4*>(p + 8);
  return f.v;
}

__device__ __forceinline__ float sigmoidf_(float x) { return 1.0f / (1.0f + __expf(-x)); }

// ---------------------------------------------------------------------------
// Elementwise helpers
// ---------------------------------------------------------------------------
__global__ void cvt_f32_bf16(const float* __restrict__ src, __bf16* __restrict__ dst, long long n) {
  long long i = (long long)blockIdx.x * blockDim.x + threadIdx.x;
  if (i < n) dst[i] = (__bf16)src[i];
}

__global__ void bias_sum(const float* __restrict__ a, const float* __restrict__ b,
                         float* __restrict__ dst, int n) {
  int i = blockIdx.x * blockDim.x + threadIdx.x;
  if (i < n) dst[i] = a[i] + b[i];
}

__global__ void scan_init(__bf16* __restrict__ hbufs, unsigned* __restrict__ cnts) {
  int i = blockIdx.x * blockDim.x + threadIdx.x;
  if (i < 4 * BB * HH) hbufs[i] = (__bf16)0.0f;
  if (i < 2) cnts[i] = 0u;
}

// Pack W (N x K fp32 row-major, N=2048) into WMMA B-fragment tile layout:
// for tile (nt,kt), lane L, elem e: value = W[nt*16 + L%16][kt*32 + e%8 + 8*(L/16) + 16*(e/8)]
__global__ void pack_w(const float* __restrict__ W, __bf16* __restrict__ dst,
                       int K, int KT, long long total) {
  long long i = (long long)blockIdx.x * blockDim.x + threadIdx.x;
  if (i >= total) return;
  int e = (int)(i & 15);
  int L = (int)((i >> 4) & 31);
  long long tile = i >> 9;
  int kt = (int)(tile % KT);
  long long nt = tile / KT;
  int n = (int)(nt * 16 + (L & 15));
  int k = kt * 32 + (e & 7) + 8 * (L >> 4) + 16 * (e >> 3);
  dst[i] = (__bf16)W[(size_t)n * K + k];
}

// ---------------------------------------------------------------------------
// Input projection GEMM: out[r][n] = sum_k A[r][k]*W[n][k] + bias[n]
// A: rows x K bf16 (rows = T*B), W packed, out fp32 rows x 2048.
// Block 256 threads (8 waves), tile 64 rows x 128 cols; grid (rows/64,16,2dirs)
// ---------------------------------------------------------------------------
__global__ __launch_bounds__(256)
void in_gemm(const __bf16* __restrict__ A,
             const __bf16* __restrict__ Wp_f, const __bf16* __restrict__ Wp_b,
             const float* __restrict__ bs_f,  const float* __restrict__ bs_b,
             float* __restrict__ out_f,       float* __restrict__ out_b,
             int K) {
  const __bf16* Wp = blockIdx.z ? Wp_b : Wp_f;
  const float*  bs = blockIdx.z ? bs_b : bs_f;
  float*        out = blockIdx.z ? out_b : out_f;
  const int KT = K >> 5;

  const int tid  = threadIdx.x;
  const int wave = tid >> 5;
  const int lane = tid & 31;
  const int half = lane >> 4;
  const int ln16 = lane & 15;
  const int mt   = wave >> 1;            // 0..3
  const int ntl  = (wave & 1) * 4;       // 0 or 4
  const int row0 = blockIdx.x * 64 + mt * 16;

  v8f acc[4];
#pragma unroll
  for (int c = 0; c < 4; ++c)
#pragma unroll
    for (int e = 0; e < 8; ++e) acc[c][e] = 0.0f;

  unsigned offB[4];
#pragma unroll
  for (int c = 0; c < 4; ++c)
    offB[c] = ((unsigned)((blockIdx.y * 8 + ntl + c) * KT) * 32u + (unsigned)lane) * 32u;

  const __bf16* arow = A + (size_t)(row0 + ln16) * K;
  for (int kt = 0; kt < KT; ++kt) {
    v16bf a = load_a_frag(arow, kt * 32, half);
#pragma unroll
    for (int c = 0; c < 4; ++c) {
      v16bf b = load_b_g(Wp, offB[c]);
      offB[c] += 1024u;  // one tile (32 lanes * 32B) per kt
      acc[c] = __builtin_amdgcn_wmma_f32_16x16x32_bf16(
          false, a, false, b, (short)0, acc[c], false, false);
    }
  }
#pragma unroll
  for (int c = 0; c < 4; ++c) {
    int ncol = blockIdx.y * 128 + (ntl + c) * 16 + ln16;
    float bias = bs[ncol];
#pragma unroll
    for (int e = 0; e < 8; ++e) {
      int m = row0 + e + 8 * half;
      out[(size_t)m * GG + ncol] = acc[c][e] + bias;
    }
  }
}

// ---------------------------------------------------------------------------
// Recurrent scan. grid = (16 j-slices, 2 directions), 256 threads (8 waves).
// Block p owns hidden cols [p*32, p*32+32) for all 4 gates: 8 n-tiles whose
// Whh slice (128KB bf16) is staged into LDS ONCE and reused for all steps.
// Wave w: mt = w&3 (batch tile), jl = w>>2 (0/1): 4 gate tiles, acc in regs,
// c in regs. h state exchanged via double-buffered global buffer + per-dir
// atomic-counter barrier (release: __threadfence; acquire: __threadfence =
// WGP$ invalidate, since WGP$ is read-only and stores land in L2).
// ---------------------------------------------------------------------------
__global__ __launch_bounds__(256)
void lstm_scan(const float* __restrict__ xg_f, const float* __restrict__ xg_b,
               const __bf16* __restrict__ whhp_f, const __bf16* __restrict__ whhp_b,
               __bf16* __restrict__ y_bf, float* __restrict__ y_f32,
               float* __restrict__ hn, float* __restrict__ cn,
               __bf16* __restrict__ hbufs, unsigned* __restrict__ cnts, int layer) {
  extern __shared__ char smem_raw[];
  __bf16* whh_lds = (__bf16*)smem_raw;   // [8 ntl][16 kt][32 lane][16 e] = 128KB

  const int p   = blockIdx.x;            // j-slice 0..15
  const int dir = blockIdx.y;            // 0..1
  const float*  xg   = dir ? xg_b : xg_f;
  const __bf16* whhp = dir ? whhp_b : whhp_f;
  unsigned* cnt = cnts + dir;
  __bf16* hb0 = hbufs + (size_t)(dir * 2 + 0) * (BB * HH);
  __bf16* hb1 = hbufs + (size_t)(dir * 2 + 1) * (BB * HH);

  const int tid  = threadIdx.x;
  const int wave = tid >> 5;
  const int lane = tid & 31;
  const int half = lane >> 4;
  const int ln16 = lane & 15;
  const int mt   = wave & 3;             // batch tile 0..3
  const int jl   = wave >> 2;            // 0..1
  const int jt   = 2 * p + jl;           // global j tile 0..31

  // Stage this block's Whh slice (gates 0..3 x local j-tiles) into LDS.
  {
    uint4* dst = (uint4*)whh_lds;
    const uint4* src = (const uint4*)whhp;
    for (int u = tid; u < 8192; u += 256) {
      int ntl = u >> 10;                 // local n-tile = g*2 + jli
      int rem = u & 1023;                // kt*64 + q
      int g = ntl >> 1, jli = ntl & 1;
      dst[u] = src[(g * 32 + 2 * p + jli) * 1024 + rem];
    }
  }

  float cfr[8], hfr[8];
#pragma unroll
  for (int e = 0; e < 8; ++e) cfr[e] = 0.0f;

  __syncthreads();

  const unsigned ldsB0 = ((unsigned)jl * 16u) * 512u + (unsigned)lane * 16u; // g=0,kt=0

  for (int s = 0; s < TT; ++s) {
    const int t = dir ? (TT - 1 - s) : s;
    const float* xg_t = xg + (size_t)t * (BB * GG);
    const __bf16* hread = (s & 1) ? hb1 : hb0;
    __bf16* hwrite = (s & 1) ? hb0 : hb1;

    // C init = input projection (+biases), fp32
    v8f acc[4];
#pragma unroll
    for (int g = 0; g < 4; ++g) {
      const int col = g * HH + jt * 16 + ln16;
#pragma unroll
      for (int e = 0; e < 8; ++e) {
        const int m = mt * 16 + e + 8 * half;
        acc[g][e] = xg_t[m * GG + col];
      }
    }

    // gates += h @ Whh^T   (A from global h buffer, B from LDS)
    const __bf16* arow = hread + (mt * 16 + ln16) * HH;
#pragma unroll 4
    for (int kt = 0; kt < HH / 32; ++kt) {
      v16bf a = load_a_frag(arow, kt * 32, half);
#pragma unroll
      for (int g = 0; g < 4; ++g) {
        v16bf b = load_b_lds(whh_lds, ldsB0 + (unsigned)(g * 2 * 16 + kt) * 512u);
        acc[g] = __builtin_amdgcn_wmma_f32_16x16x32_bf16(
            false, a, false, b, (short)0, acc[g], false, false);
      }
    }

    // gate order i,f,g,o (torch LSTMCell)
#pragma unroll
    for (int e = 0; e < 8; ++e) {
      const float ig = sigmoidf_(acc[0][e]);
      const float fg = sigmoidf_(acc[1][e]);
      const float gg = tanhf(acc[2][e]);
      const float og = sigmoidf_(acc[3][e]);
      const float c  = fg * cfr[e] + ig * gg;
      cfr[e] = c;
      hfr[e] = og * tanhf(c);
    }

    // write h slice (next-step buffer) + layer output
    const int j = jt * 16 + ln16;
#pragma unroll
    for (int e = 0; e < 8; ++e) {
      const int m = mt * 16 + e + 8 * half;
      const __bf16 hb = (__bf16)hfr[e];
      hwrite[m * HH + j] = hb;
      const size_t oidx = ((size_t)t * BB + m) * (2 * HH) + (size_t)dir * HH + j;
      if (layer == 0) y_bf[oidx] = hb;
      else            y_f32[oidx] = hfr[e];
    }

    // inter-block barrier for this direction (release/acquire via L2)
    __threadfence();                     // each wave: own stores visible at L2
    __syncthreads();
    if (tid == 0) {
      atomicAdd(cnt, 1u);
      const unsigned tgt = 16u * (unsigned)(s + 1);
      while (__hip_atomic_load(cnt, __ATOMIC_RELAXED, __HIP_MEMORY_SCOPE_AGENT) < tgt)
        __builtin_amdgcn_s_sleep(2);
      __threadfence();                   // invalidate WGP$ before next reads
    }
    __syncthreads();
  }

  // final h_n / c_n (fp32)
  const int slot = layer * 2 + dir;
  float* hdst = hn + (size_t)slot * BB * HH;
  float* cdst = cn + (size_t)slot * BB * HH;
  const int j = jt * 16 + ln16;
#pragma unroll
  for (int e = 0; e < 8; ++e) {
    const int m = mt * 16 + e + 8 * half;
    hdst[m * HH + j] = hfr[e];
    cdst[m * HH + j] = cfr[e];
  }
}

// ---------------------------------------------------------------------------
// Host launcher
// ---------------------------------------------------------------------------
extern "C" void kernel_launch(void* const* d_in, const int* in_sizes, int n_in,
                              void* d_out, int out_size, void* d_ws, size_t ws_size,
                              hipStream_t stream) {
  (void)in_sizes; (void)n_in; (void)out_size; (void)ws_size;

  const float* x       = (const float*)d_in[0];
  const float* W_ih_0f = (const float*)d_in[1];
  const float* W_hh_0f = (const float*)d_in[2];
  const float* b_ih_0f = (const float*)d_in[3];
  const float* b_hh_0f = (const float*)d_in[4];
  const float* W_ih_0b = (const float*)d_in[5];
  const float* W_hh_0b = (const float*)d_in[6];
  const float* b_ih_0b = (const float*)d_in[7];
  const float* b_hh_0b = (const float*)d_in[8];
  const float* W_ih_1f = (const float*)d_in[9];
  const float* W_hh_1f = (const float*)d_in[10];
  const float* b_ih_1f = (const float*)d_in[11];
  const float* b_hh_1f = (const float*)d_in[12];
  const float* W_ih_1b = (const float*)d_in[13];
  const float* W_hh_1b = (const float*)d_in[14];
  const float* b_ih_1b = (const float*)d_in[15];
  const float* b_hh_1b = (const float*)d_in[16];

  char* ws = (char*)d_ws;
  const size_t SZ_XBF    = (size_t)TT * BB * 512 * 2;   //  32 MB
  const size_t SZ_WP512  = (size_t)GG * 512 * 2;        //   2 MB
  const size_t SZ_WP1024 = (size_t)GG * 1024 * 2;       //   4 MB
  const size_t SZ_BS     = (size_t)GG * 4;
  const size_t SZ_Y0     = (size_t)TT * BB * 1024 * 2;  //  64 MB
  const size_t SZ_HBUF   = (size_t)4 * BB * HH * 2;     // 256 KB (2 dirs x 2 bufs)
  const size_t SZ_CNT    = 256;
  const size_t SZ_XG     = (size_t)TT * BB * GG * 4;    // 256 MB per direction

  size_t off = 0;
  __bf16*   xbf     = (__bf16*)(ws + off);   off += SZ_XBF;
  __bf16*   wih0f_p = (__bf16*)(ws + off);   off += SZ_WP512;
  __bf16*   wih0b_p = (__bf16*)(ws + off);   off += SZ_WP512;
  __bf16*   wih1f_p = (__bf16*)(ws + off);   off += SZ_WP1024;
  __bf16*   wih1b_p = (__bf16*)(ws + off);   off += SZ_WP1024;
  __bf16*   whh0f_p = (__bf16*)(ws + off);   off += SZ_WP512;
  __bf16*   whh0b_p = (__bf16*)(ws + off);   off += SZ_WP512;
  __bf16*   whh1f_p = (__bf16*)(ws + off);   off += SZ_WP512;
  __bf16*   whh1b_p = (__bf16*)(ws + off);   off += SZ_WP512;
  float*    bs0f    = (float*)(ws + off);    off += SZ_BS;
  float*    bs0b    = (float*)(ws + off);    off += SZ_BS;
  float*    bs1f    = (float*)(ws + off);    off += SZ_BS;
  float*    bs1b    = (float*)(ws + off);    off += SZ_BS;
  __bf16*   y0      = (__bf16*)(ws + off);   off += SZ_Y0;
  __bf16*   hbufs   = (__bf16*)(ws + off);   off += SZ_HBUF;
  unsigned* cnts    = (unsigned*)(ws + off); off += SZ_CNT;
  float*    xg_fwd  = (float*)(ws + off);    off += SZ_XG;
  float*    xg_bwd  = (float*)(ws + off);    off += SZ_XG;

  float* y_out = (float*)d_out;
  float* hn    = y_out + (size_t)TT * BB * 2 * HH;
  float* cn    = hn + (size_t)4 * BB * HH;

  // allow 128KB dynamic LDS for the scan kernel (CDNA5 WGP has 320KB)
  hipFuncSetAttribute((const void*)lstm_scan,
                      hipFuncAttributeMaxDynamicSharedMemorySize, 131072);

  // 1) x -> bf16
  {
    long long n = (long long)TT * BB * 512;
    cvt_f32_bf16<<<dim3((unsigned)((n + 255) / 256)), dim3(256), 0, stream>>>(x, xbf, n);
  }
  // 2) pack weights into WMMA B-tile layout
  {
    long long t512  = (long long)GG * 512;
    long long t1024 = (long long)GG * 1024;
    dim3 b(256);
    dim3 g512((unsigned)((t512 + 255) / 256));
    dim3 g1024((unsigned)((t1024 + 255) / 256));
    pack_w<<<g512,  b, 0, stream>>>(W_ih_0f, wih0f_p, 512,  16, t512);
    pack_w<<<g512,  b, 0, stream>>>(W_ih_0b, wih0b_p, 512,  16, t512);
    pack_w<<<g1024, b, 0, stream>>>(W_ih_1f, wih1f_p, 1024, 32, t1024);
    pack_w<<<g1024, b, 0, stream>>>(W_ih_1b, wih1b_p, 1024, 32, t1024);
    pack_w<<<g512,  b, 0, stream>>>(W_hh_0f, whh0f_p, 512,  16, t512);
    pack_w<<<g512,  b, 0, stream>>>(W_hh_0b, whh0b_p, 512,  16, t512);
    pack_w<<<g512,  b, 0, stream>>>(W_hh_1f, whh1f_p, 512,  16, t512);
    pack_w<<<g512,  b, 0, stream>>>(W_hh_1b, whh1b_p, 512,  16, t512);
  }
  // 3) bias sums
  {
    dim3 g(2), b(1024);
    bias_sum<<<g, b, 0, stream>>>(b_ih_0f, b_hh_0f, bs0f, GG);
    bias_sum<<<g, b, 0, stream>>>(b_ih_0b, b_hh_0b, bs0b, GG);
    bias_sum<<<g, b, 0, stream>>>(b_ih_1f, b_hh_1f, bs1f, GG);
    bias_sum<<<g, b, 0, stream>>>(b_ih_1b, b_hh_1b, bs1b, GG);
  }
  // 4) layer0 input projections (both directions)
  {
    dim3 g(TT * BB / 64, GG / 128, 2), b(256);
    in_gemm<<<g, b, 0, stream>>>(xbf, wih0f_p, wih0b_p, bs0f, bs0b, xg_fwd, xg_bwd, 512);
  }
  // 5) layer0 scan (fwd+bwd across 32 WGPs), y0 in bf16
  scan_init<<<dim3(512), dim3(256), 0, stream>>>(hbufs, cnts);
  lstm_scan<<<dim3(16, 2), dim3(256), 131072, stream>>>(
      xg_fwd, xg_bwd, whh0f_p, whh0b_p, y0, y_out, hn, cn, hbufs, cnts, 0);
  // 6) layer1 input projections from y0 (K=1024)
  {
    dim3 g(TT * BB / 64, GG / 128, 2), b(256);
    in_gemm<<<g, b, 0, stream>>>(y0, wih1f_p, wih1b_p, bs1f, bs1b, xg_fwd, xg_bwd, 1024);
  }
  // 7) layer1 scan, fp32 output straight into d_out
  scan_init<<<dim3(512), dim3(256), 0, stream>>>(hbufs, cnts);
  lstm_scan<<<dim3(16, 2), dim3(256), 131072, stream>>>(
      xg_fwd, xg_bwd, whh1f_p, whh1b_p, y0, y_out, hn, cn, hbufs, cnts, 1);
}
